// DecoderLayer_78039555768395
// MI455X (gfx1250) — matrix operand order repaired and checked by reading, not compiled, not run on hardware
//
#include <hip/hip_runtime.h>

#define S_LEN 2048
#define D_DIM 2048
#define H_NUM 16
#define HD_DIM 128
#define DFF_DIM 8192
#define EPSF 1e-6f
#define THETAF 10000.0f
#define KPRUNE 204            // int(0.1 * 2048)
#define SKEEP (S_LEN - KPRUNE)

#if defined(__has_builtin)
#if __has_builtin(__builtin_amdgcn_global_load_async_to_lds_b128) && \
    __has_builtin(__builtin_amdgcn_s_wait_asynccnt)
#define USE_ASYNC_LDS 1
#endif
#endif

typedef __attribute__((ext_vector_type(16))) __bf16 v16bf;
typedef __attribute__((ext_vector_type(8)))  float  v8f;
typedef int v4i __attribute__((vector_size(16)));   // matches builtin param type
typedef __attribute__((address_space(1))) v4i* gv4i_ptr;
typedef __attribute__((address_space(3))) v4i* lv4i_ptr;

__device__ __forceinline__ unsigned short f2bf(float f) {
  union { float f; unsigned u; } x; x.f = f;
  unsigned u = x.u;
  u += 0x7FFFu + ((u >> 16) & 1u);   // round-to-nearest-even
  return (unsigned short)(u >> 16);
}

// ---------------------------------------------------------------------------
// bf16-in / f32-accumulate WMMA GEMM.
//   C[z] = scale * (A[z] x B[z]) (+ Res)    A: MxK row-major, B: KxN row-major
//   BT=true reads B as B[n*ldb + k] (B stored NxK, used transposed: Q*K^T)
// Block tile 128(M) x 128(N) x 32(K); 512 threads = 16 waves (4x4), each wave
// owns a 32x32 strip -> 4 v_wmma_f32_16x16x32_bf16 per K-step with A/B reuse.
// A tile staged raw f32 via GLOBAL_LOAD_ASYNC_TO_LDS_B128 (ASYNCcnt path),
// converted to bf16 at fragment build; B tile converted+transposed via VALU.
// ---------------------------------------------------------------------------
template <bool BT>
__global__ __launch_bounds__(512)
void gemm_bf16_wmma(const float* __restrict__ A, int lda, long long sA,
                    const float* __restrict__ B, int ldb, long long sB,
                    float* __restrict__ C, int ldc, long long sC,
                    const float* __restrict__ Res, int ldres,
                    int M, int N, int K, float scale) {
  __shared__ __attribute__((aligned(16))) float lAf[128 * 32];   // f32 [m][k]
  __shared__ unsigned short lB[128 * 32];                        // bf16 [n][k]

  const int z = blockIdx.z;
  A += (long long)z * sA;
  B += (long long)z * sB;
  C += (long long)z * sC;

  const int n0 = blockIdx.x * 128;
  const int m0 = blockIdx.y * 128;
  const int tid  = threadIdx.x;
  const int wave = tid >> 5;
  const int lane = tid & 31;
  const int wr = wave & 3;        // wave M-row (0..3) -> 32 rows each
  const int wc = wave >> 2;       // wave N-col (0..3) -> 32 cols each
  const int half = lane >> 4;     // lane half per ISA fragment layout
  const int r    = lane & 15;

  v8f acc00 = {}, acc01 = {}, acc10 = {}, acc11 = {};

  for (int k0 = 0; k0 < K; k0 += 32) {
    // ---- stage A tile (128x32 f32) ----
#if USE_ASYNC_LDS
    for (int j = 0; j < 2; ++j) {
      int c   = tid + j * 512;         // 1024 chunks of 16B
      int row = c >> 3;                // 8 chunks per 32-float row
      int col = (c & 7) * 4;
      const float* src = A + (long long)(m0 + row) * lda + (k0 + col);
      __builtin_amdgcn_global_load_async_to_lds_b128(
          (gv4i_ptr)(src),
          (lv4i_ptr)(&lAf[row * 32 + col]),
          0, 0);
    }
#else
    for (int j = 0; j < 8; ++j) {
      int idx = tid * 8 + j;
      int row = idx >> 5, col = idx & 31;
      lAf[idx] = A[(long long)(m0 + row) * lda + (k0 + col)];
    }
#endif
    // ---- stage B tile (32x128) as bf16, transposed to [n][k] ----
    for (int j = 0; j < 8; ++j) {
      int idx = tid * 8 + j;
      int n = idx >> 5, kk = idx & 31;
      float bv = BT ? B[(long long)(n0 + n) * ldb + (k0 + kk)]
                    : B[(long long)(k0 + kk) * ldb + (n0 + n)];
      lB[n * 32 + kk] = f2bf(bv);
    }
#if USE_ASYNC_LDS
    __builtin_amdgcn_s_wait_asynccnt(0);
#endif
    __syncthreads();

    // ---- build fragments per ISA VGPR layout ----
    __attribute__((aligned(32))) unsigned short a0f[16], a1f[16], b0f[16], b1f[16];
    const int aRow0 = wr * 32 + r;
    const int aRow1 = aRow0 + 16;
    for (int j = 0; j < 8; ++j) {
      a0f[j]     = f2bf(lAf[aRow0 * 32 +      half * 8 + j]);  // K = half*8+j
      a0f[8 + j] = f2bf(lAf[aRow0 * 32 + 16 + half * 8 + j]);  // K = 16+half*8+j
      a1f[j]     = f2bf(lAf[aRow1 * 32 +      half * 8 + j]);
      a1f[8 + j] = f2bf(lAf[aRow1 * 32 + 16 + half * 8 + j]);
    }
    const int c0 = wc * 32 + r;
    const int c1 = c0 + 16;
    for (int j = 0; j < 8; ++j) {
      b0f[j]     = lB[c0 * 32 +      half * 8 + j];            // K = half*8+j
      b0f[8 + j] = lB[c0 * 32 + 16 + half * 8 + j];            // K = 16+half*8+j
      b1f[j]     = lB[c1 * 32 +      half * 8 + j];
      b1f[8 + j] = lB[c1 * 32 + 16 + half * 8 + j];
    }
    v16bf a0 = *reinterpret_cast<v16bf*>(a0f);
    v16bf a1 = *reinterpret_cast<v16bf*>(a1f);
    v16bf b0 = *reinterpret_cast<v16bf*>(b0f);
    v16bf b1 = *reinterpret_cast<v16bf*>(b1f);

    acc00 = __builtin_amdgcn_wmma_f32_16x16x32_bf16(false, a0, false, b0,
                                                    (short)0, acc00, false, false);
    acc01 = __builtin_amdgcn_wmma_f32_16x16x32_bf16(false, a0, false, b1,
                                                    (short)0, acc01, false, false);
    acc10 = __builtin_amdgcn_wmma_f32_16x16x32_bf16(false, a1, false, b0,
                                                    (short)0, acc10, false, false);
    acc11 = __builtin_amdgcn_wmma_f32_16x16x32_bf16(false, a1, false, b1,
                                                    (short)0, acc11, false, false);
    __syncthreads();
  }

  // ---- epilogue: scale + optional residual ----
  for (int j = 0; j < 8; ++j) {
    long long row0 = m0 + wr * 32 + half * 8 + j;   // C/D: M = j + 8*half
    long long row1 = row0 + 16;
    long long col0 = n0 + wc * 32 + r;
    long long col1 = col0 + 16;
    float v00 = acc00[j] * scale, v01 = acc01[j] * scale;
    float v10 = acc10[j] * scale, v11 = acc11[j] * scale;
    if (Res) {
      v00 += Res[row0 * ldres + col0];
      v01 += Res[row0 * ldres + col1];
      v10 += Res[row1 * ldres + col0];
      v11 += Res[row1 * ldres + col1];
    }
    C[row0 * ldc + col0] = v00;
    C[row0 * ldc + col1] = v01;
    C[row1 * ldc + col0] = v10;
    C[row1 * ldc + col1] = v11;
  }
}

// ---------------------------------------------------------------------------
__global__ __launch_bounds__(256)
void rmsnorm_kernel(const float* __restrict__ x, const float* __restrict__ w,
                    float* __restrict__ out, int D) {
  __shared__ float red[256];
  const int row = blockIdx.x, tid = threadIdx.x;
  const float* xr = x + (long long)row * D;
  float s = 0.f;
  for (int c = tid; c < D; c += 256) { float v = xr[c]; s += v * v; }
  red[tid] = s; __syncthreads();
  for (int o = 128; o > 0; o >>= 1) {
    if (tid < o) red[tid] += red[tid + o];
    __syncthreads();
  }
  const float rs = rsqrtf(red[0] / (float)D + EPSF);
  float* orow = out + (long long)row * D;
  for (int c = tid; c < D; c += 256) orow[c] = xr[c] * rs * w[c];
}

__global__ __launch_bounds__(256)
void rope_kernel(float* __restrict__ x, const int* __restrict__ pos) {
  const int idx = blockIdx.x * 256 + threadIdx.x;
  const int halfd = HD_DIM / 2;
  const int d = idx % halfd;
  const int t = idx / halfd;
  const int h = t % H_NUM;
  const int s = t / H_NUM;
  if (s >= S_LEN) return;
  float inv = __powf(THETAF, -(float)d / (float)halfd);
  float ang = (float)pos[s] * inv;
  float cs = __cosf(ang), sn = __sinf(ang);
  float* base = x + (long long)s * D_DIM + h * HD_DIM;
  float x1 = base[d], x2 = base[d + halfd];
  base[d]         = x1 * cs - x2 * sn;
  base[d + halfd] = x2 * cs + x1 * sn;
}

__global__ __launch_bounds__(256)
void softmax_kernel(float* __restrict__ attn, const float* __restrict__ mask) {
  __shared__ float red[256];
  const int row = blockIdx.x, h = blockIdx.y, tid = threadIdx.x;
  float* p = attn + ((long long)h * S_LEN + row) * S_LEN;
  const float* mr = mask + (long long)row * S_LEN;
  float m = -3.4e38f;
  for (int c = tid; c < S_LEN; c += 256) m = fmaxf(m, p[c] + mr[c]);
  red[tid] = m; __syncthreads();
  for (int o = 128; o > 0; o >>= 1) {
    if (tid < o) red[tid] = fmaxf(red[tid], red[tid + o]);
    __syncthreads();
  }
  m = red[0]; __syncthreads();
  float s = 0.f;
  for (int c = tid; c < S_LEN; c += 256) {
    float e = __expf(p[c] + mr[c] - m);
    p[c] = e; s += e;
  }
  red[tid] = s; __syncthreads();
  for (int o = 128; o > 0; o >>= 1) {
    if (tid < o) red[tid] += red[tid + o];
    __syncthreads();
  }
  const float inv = 1.f / red[0];
  for (int c = tid; c < S_LEN; c += 256) p[c] *= inv;
}

__global__ __launch_bounds__(256)
void silu_mul_kernel(float* __restrict__ gate, const float* __restrict__ up,
                     long long n) {
  long long i = (long long)blockIdx.x * 256 + threadIdx.x;
  if (i < n) {
    float g = gate[i];
    gate[i] = g / (1.f + __expf(-g)) * up[i];
  }
}

__global__ __launch_bounds__(256)
void importance_kernel(const float* __restrict__ attn, float* __restrict__ imp) {
  const int c = blockIdx.x * 256 + threadIdx.x;
  if (c >= S_LEN) return;
  float s = 0.f;
  for (int h = 0; h < H_NUM; ++h)
    s += attn[(long long)h * S_LEN * S_LEN + (long long)(S_LEN - 1) * S_LEN + c];
  imp[c] = s / (float)H_NUM;
}

// 204 rounds of block-wide argmin (value, then lowest index) == top_k(-imp)
__global__ __launch_bounds__(256)
void select_prune_kernel(const float* __restrict__ imp,
                         int* __restrict__ pruned_g,
                         int* __restrict__ new_selected) {
  __shared__ float sv[256];
  __shared__ int   si[256];
  __shared__ unsigned char flag[S_LEN];
  const int tid = threadIdx.x;
  for (int c = tid; c < S_LEN; c += 256) flag[c] = 0;
  __syncthreads();
  for (int round = 0; round < KPRUNE; ++round) {
    float bv = 3.4e38f; int bi = S_LEN;
    for (int c = tid; c < S_LEN; c += 256) {
      if (!flag[c]) {
        float v = imp[c];
        if (v < bv || (v == bv && c < bi)) { bv = v; bi = c; }
      }
    }
    sv[tid] = bv; si[tid] = bi; __syncthreads();
    for (int o = 128; o > 0; o >>= 1) {
      if (tid < o) {
        if (sv[tid + o] < sv[tid] ||
            (sv[tid + o] == sv[tid] && si[tid + o] < si[tid])) {
          sv[tid] = sv[tid + o]; si[tid] = si[tid + o];
        }
      }
      __syncthreads();
    }
    if (tid == 0) flag[si[0]] = 1;
    __syncthreads();
  }
  for (int c = tid; c < S_LEN; c += 256) {
    pruned_g[c]     = flag[c];
    new_selected[c] = flag[c] ? 0 : 1;
  }
}

__global__ void scan_keep_kernel(const int* __restrict__ pruned,
                                 int* __restrict__ keep) {
  if (threadIdx.x == 0 && blockIdx.x == 0) {
    int cnt = 0;
    for (int s = 0; s < S_LEN; ++s)
      if (!pruned[s]) { if (cnt < SKEEP) keep[cnt] = s; ++cnt; }
  }
}

__global__ __launch_bounds__(256)
void gather_kernel(const float* __restrict__ newh, const int* __restrict__ keep,
                   float* __restrict__ out_h) {
  const int row = blockIdx.x;
  const int c = blockIdx.y * 256 + threadIdx.x;
  out_h[(long long)row * D_DIM + c] = newh[(long long)keep[row] * D_DIM + c];
}

// ---------------------------------------------------------------------------
extern "C" void kernel_launch(void* const* d_in, const int* in_sizes, int n_in,
                              void* d_out, int out_size, void* d_ws, size_t ws_size,
                              hipStream_t stream) {
  const float* x    = (const float*)d_in[0];
  const float* mask = (const float*)d_in[1];
  const float* n1w  = (const float*)d_in[2];
  const float* n2w  = (const float*)d_in[3];
  const float* Wq   = (const float*)d_in[4];
  const float* Wk   = (const float*)d_in[5];
  const float* Wv   = (const float*)d_in[6];
  const float* Wo   = (const float*)d_in[7];
  const float* Wg   = (const float*)d_in[8];
  const float* Wu   = (const float*)d_in[9];
  const float* Wd   = (const float*)d_in[10];
  const int*   pos  = (const int*)d_in[12];

  float* ws = (float*)d_ws;
  const long long SD = (long long)S_LEN * D_DIM;
  float* hn   = ws;              // rmsnorm1 out; later reused for new_h
  float* q    = ws + SD;         // Q; later reused for ctx
  float* kbuf = ws + 2 * SD;     // K; later reused for h2
  float* vbuf = ws + 3 * SD;     // V; later reused for g (rmsnorm2 out)
  float* gate = ws + 4 * SD;     // S x DFF (later SiLU(gate)*up in place)
  float* up   = gate + (long long)S_LEN * DFF_DIM;
  float* imp  = up + (long long)S_LEN * DFF_DIM;
  int* pruned = (int*)(imp + S_LEN);

  // outputs, concatenated flat: out_h, new_selected, keep_idxs, attn
  float* out_h = (float*)d_out;
  int* nsel = (int*)d_out + (long long)SKEEP * D_DIM;
  int* keep = nsel + S_LEN;
  float* attn = (float*)d_out + (long long)SKEEP * D_DIM + S_LEN + SKEEP;

  dim3 blk(256);
  dim3 blkG(512);

  // 1) h = rmsnorm(x, norm1_w)
  rmsnorm_kernel<<<S_LEN, blk, 0, stream>>>(x, n1w, hn, D_DIM);

  // 2) Q/K/V projections
  dim3 gQ(D_DIM / 128, S_LEN / 128, 1);
  gemm_bf16_wmma<false><<<gQ, blkG, 0, stream>>>(hn, D_DIM, 0, Wq, D_DIM, 0,
      q, D_DIM, 0, nullptr, 0, S_LEN, D_DIM, D_DIM, 1.0f);
  gemm_bf16_wmma<false><<<gQ, blkG, 0, stream>>>(hn, D_DIM, 0, Wk, D_DIM, 0,
      kbuf, D_DIM, 0, nullptr, 0, S_LEN, D_DIM, D_DIM, 1.0f);
  gemm_bf16_wmma<false><<<gQ, blkG, 0, stream>>>(hn, D_DIM, 0, Wv, D_DIM, 0,
      vbuf, D_DIM, 0, nullptr, 0, S_LEN, D_DIM, D_DIM, 1.0f);

  // 3) RoPE on Q, K
  const int ropeBlocks = (S_LEN * H_NUM * (HD_DIM / 2)) / 256;
  rope_kernel<<<ropeBlocks, blk, 0, stream>>>(q, pos);
  rope_kernel<<<ropeBlocks, blk, 0, stream>>>(kbuf, pos);

  // 4) scores = (Q K^T) / sqrt(HD), batched over heads (B transposed)
  dim3 gS(S_LEN / 128, S_LEN / 128, H_NUM);
  gemm_bf16_wmma<true><<<gS, blkG, 0, stream>>>(q, D_DIM, HD_DIM,
      kbuf, D_DIM, HD_DIM, attn, S_LEN, (long long)S_LEN * S_LEN,
      nullptr, 0, S_LEN, S_LEN, HD_DIM, 0.08838834764831845f);

  // 5) softmax(scores + causal_mask) in place
  dim3 gSm(S_LEN, H_NUM);
  softmax_kernel<<<gSm, blk, 0, stream>>>(attn, mask);

  // 6) ctx = attn @ V (batched per head) -> reuse q buffer
  float* ctx = q;
  dim3 gC(HD_DIM / 128, S_LEN / 128, H_NUM);
  gemm_bf16_wmma<false><<<gC, blkG, 0, stream>>>(attn, S_LEN,
      (long long)S_LEN * S_LEN, vbuf, D_DIM, HD_DIM, ctx, D_DIM, HD_DIM,
      nullptr, 0, S_LEN, HD_DIM, S_LEN, 1.0f);

  // 7) h2 = x + ctx @ Wo  -> reuse kbuf
  float* h2 = kbuf;
  gemm_bf16_wmma<false><<<gQ, blkG, 0, stream>>>(ctx, D_DIM, 0, Wo, D_DIM, 0,
      h2, D_DIM, 0, x, D_DIM, S_LEN, D_DIM, D_DIM, 1.0f);

  // 8) g = rmsnorm(h2, norm2_w) -> reuse vbuf
  float* g = vbuf;
  rmsnorm_kernel<<<S_LEN, blk, 0, stream>>>(h2, n2w, g, D_DIM);

  // 9) gate = g @ Wgate ; up = g @ Wup
  dim3 gF(DFF_DIM / 128, S_LEN / 128, 1);
  gemm_bf16_wmma<false><<<gF, blkG, 0, stream>>>(g, D_DIM, 0, Wg, DFF_DIM, 0,
      gate, DFF_DIM, 0, nullptr, 0, S_LEN, DFF_DIM, D_DIM, 1.0f);
  gemm_bf16_wmma<false><<<gF, blkG, 0, stream>>>(g, D_DIM, 0, Wu, DFF_DIM, 0,
      up, DFF_DIM, 0, nullptr, 0, S_LEN, DFF_DIM, D_DIM, 1.0f);

  // 10) act = silu(gate) * up (in place in gate)
  const long long nAct = (long long)S_LEN * DFF_DIM;
  silu_mul_kernel<<<(unsigned)(nAct / 256), blk, 0, stream>>>(gate, up, nAct);

  // 11) new_h = h2 + act @ Wdown -> reuse hn
  float* newh = hn;
  gemm_bf16_wmma<false><<<gQ, blkG, 0, stream>>>(gate, DFF_DIM, 0, Wd, D_DIM, 0,
      newh, D_DIM, 0, h2, D_DIM, S_LEN, D_DIM, DFF_DIM, 1.0f);

  // 12) importance = mean over heads of attn[:, :, -1, :]
  importance_kernel<<<S_LEN / 256, blk, 0, stream>>>(attn, imp);

  // 13) prune 204 least-important positions; write new_selected
  select_prune_kernel<<<1, blk, 0, stream>>>(imp, pruned, nsel);

  // 14) keep_idxs = indices of kept positions, in order
  scan_keep_kernel<<<1, blk, 0, stream>>>(pruned, keep);

  // 15) out_h = new_h[keep_idxs, :]
  dim3 gG(SKEEP, D_DIM / 256);
  gather_kernel<<<gG, blk, 0, stream>>>(newh, keep, out_h);
}
